// MSGDC_47699906789724
// MI455X (gfx1250) — matrix operand order repaired and checked by reading, not compiled, not run on hardware
//
#include <hip/hip_runtime.h>
#include <math.h>

// ---------- types ----------
typedef __attribute__((ext_vector_type(16))) __bf16   v16bf;
typedef __attribute__((ext_vector_type(8)))  float    v8f;
typedef __attribute__((ext_vector_type(4)))  uint32_t u32x4;

union BF2   { __bf16 h[2];  uint32_t u; };
union AFrag { v16bf v; u32x4 q[2]; __bf16 h[16]; };

// ---------- CDNA5 async global->LDS (guarded; falls back to load+ds_store) ----------
#if defined(__AMDGCN__) && __has_builtin(__builtin_amdgcn_global_load_async_to_lds_b32)
  #define HAS_ASYNC_LDS 1
#else
  #define HAS_ASYNC_LDS 0
#endif

#if defined(__AMDGCN__) && __has_builtin(__builtin_amdgcn_s_wait_asynccnt)
  #define WAIT_ASYNC() __builtin_amdgcn_s_wait_asynccnt(0)
#elif HAS_ASYNC_LDS
  #define WAIT_ASYNC() asm volatile("s_wait_asynccnt 0" ::: "memory")
#else
  #define WAIT_ASYNC()
#endif

// builtin expects int* params (per hipcc diagnostic); AS-qualified pointers
// match exactly if params are AS-qualified, else implicitly decay to generic.
#define GLOBAL_AS(p) ((__attribute__((address_space(1))) int*)(p))
#define LDS_AS(p)    ((__attribute__((address_space(3))) int*)(p))

// ---------- problem constants ----------
#define B_    4
#define C_    64
#define H_    128
#define W_    128
#define KK_   9
#define OFFC_ 54
#define NPIX  (B_ * H_ * W_)          // 65536
#define NTILE (NPIX / 16)             // 4096 tiles of 16 pixels (16 consecutive w)

#define AS_STRIDE 584   // 576 + 8 pad (bf16) -> 1168B row, 16B aligned, conflict-free frag loads
#define HS_STRIDE 136   // 128 + 8 pad (bf16)

// ---------- WMMA helpers ----------
__device__ __forceinline__ v8f wmma_bf16(v16bf a, v16bf b, v8f c) {
  return __builtin_amdgcn_wmma_f32_16x16x32_bf16(false, a, false, b, (short)0, c, false, false);
}

// B fragment: packed by pack_b_kernel so each lane reads 32 contiguous bytes.
__device__ __forceinline__ v16bf load_b_frag(const uint32_t* __restrict__ bp,
                                             int ks, int nt, int ntiles, int lane) {
  const uint32_t* p = bp + ((size_t)(ks * ntiles + nt) * 32 + (size_t)lane) * 8;
  AFrag u;
  u.q[0] = *(const u32x4*)p;
  u.q[1] = *(const u32x4*)(p + 4);
  return u.v;
}

// A fragment from an LDS tile: lanes 0-15 -> rows m, K {0..7}+{16..23}; lanes 16-31 -> K {8..15}+{24..31}
__device__ __forceinline__ v16bf load_a_frag_lds(const __bf16* As, int stride, int ks, int lane) {
  int m = lane & 15, half = lane >> 4;
  const __bf16* row = As + m * stride + ks * 32 + half * 8;
  AFrag u;
  u.q[0] = *(const u32x4*)row;
  u.q[1] = *(const u32x4*)(row + 16);
  return u.v;
}

// ---------- weight packing into B-fragment order ----------
// mode 0: conv weight src(O, C_, 3, 3), B[k][n] = src[n][k&63][k>>6] (K order = kk*64+c), zero-pad n>=Osrc
// mode 1: dense src(N, Ktot) row-major,  B[k][n] = src[n*Ktot + k]
__global__ void pack_b_kernel(const float* __restrict__ src, uint32_t* __restrict__ dst,
                              int mode, int Ktot, int Osrc, int ksteps, int ntiles) {
  int idx   = blockIdx.x * blockDim.x + threadIdx.x;
  int total = ksteps * ntiles * 32 * 8;
  if (idx >= total) return;
  int v    = idx & 7;
  int lane = (idx >> 3) & 31;
  int nt   = (idx >> 8) % ntiles;
  int ks   = (idx >> 8) / ntiles;
  int n    = nt * 16 + (lane & 15);
  int k0   = ks * 32 + (v >> 2) * 16 + (lane >> 4) * 8 + (v & 3) * 2;
  float f0 = 0.f, f1 = 0.f;
  if (mode == 0) {
    if (n < Osrc) {
      if (k0     < Ktot) f0 = src[((size_t)n * C_ + (k0 & 63)) * KK_ + (k0 >> 6)];
      if (k0 + 1 < Ktot) f1 = src[((size_t)n * C_ + ((k0 + 1) & 63)) * KK_ + ((k0 + 1) >> 6)];
    }
  } else {
    if (k0     < Ktot) f0 = src[(size_t)n * Ktot + k0];
    if (k0 + 1 < Ktot) f1 = src[(size_t)n * Ktot + k0 + 1];
  }
  BF2 pk; pk.h[0] = (__bf16)f0; pk.h[1] = (__bf16)f1;
  dst[idx] = pk.u;
}

// ---------- stage 1: BN + SiLU residual ----------
__global__ void bn_silu_kernel(const float* __restrict__ x,
                               const float* __restrict__ gamma, const float* __restrict__ beta,
                               const float* __restrict__ mean,  const float* __restrict__ var,
                               const float* __restrict__ res_scale,
                               float* __restrict__ out0, __bf16* __restrict__ out0h) {
  int idx = blockIdx.x * 256 + threadIdx.x;
  if (idx >= NPIX * C_) return;
  int c = (idx >> 14) & 63;
  float xv = x[idx];
  float nm = (xv - mean[c]) * (gamma[c] * rsqrtf(var[c] + 1e-5f)) + beta[c];
  float o  = xv + res_scale[0] * (nm / (1.f + __expf(-nm)));
  out0[idx] = o;
  int w = idx & 127, h = (idx >> 7) & 127, b = idx >> 20;
  out0h[((((size_t)b * H_) + h) * W_ + w) * C_ + c] = (__bf16)o;
}

// ---------- stage 2: 3x3 offset conv (implicit GEMM, K=576, N=54 padded to 64) ----------
__global__ __launch_bounds__(32) void offset_conv_kernel(
    const __bf16* __restrict__ xh, const uint32_t* __restrict__ Bp,
    const float* __restrict__ off_b, float* __restrict__ offs) {
  __shared__ __bf16 As[16 * AS_STRIDE];
  int tile = blockIdx.x;
  int wb = (tile & 7) * 16, h = (tile >> 3) & 127, b = tile >> 10;
  int lane = threadIdx.x;
  __builtin_prefetch(Bp, 0, 3);
  // A-tile build is a pure copy: use CDNA5 async global->LDS when available.
  for (int m = 0; m < 16; ++m) {
    int w = wb + m;
    #pragma unroll
    for (int kk = 0; kk < KK_; ++kk) {
      int iy = h + (kk / 3) - 1, ix = w + (kk % 3) - 1;
      __bf16* ldst = As + m * AS_STRIDE + kk * 64 + lane * 2;
      if (iy >= 0 && iy < H_ && ix >= 0 && ix < W_) {   // uniform across lanes
        const __bf16* gsrc = xh + (((size_t)b * H_ + iy) * W_ + ix) * C_ + lane * 2;
#if HAS_ASYNC_LDS
        __builtin_amdgcn_global_load_async_to_lds_b32(GLOBAL_AS(gsrc), LDS_AS(ldst), 0, 0);
#else
        *(uint32_t*)ldst = *(const uint32_t*)gsrc;
#endif
      } else {
        *(uint32_t*)ldst = 0u;   // disjoint from async targets
      }
    }
  }
  WAIT_ASYNC();
  __syncthreads();
  v8f acc[4] = {};
  for (int ks = 0; ks < 18; ++ks) {
    v16bf a = load_a_frag_lds(As, AS_STRIDE, ks, lane);
    #pragma unroll
    for (int nt = 0; nt < 4; ++nt)
      acc[nt] = wmma_bf16(a, load_b_frag(Bp, ks, nt, 4, lane), acc[nt]);
  }
  int nloc = lane & 15, hm = lane >> 4;
  #pragma unroll
  for (int nt = 0; nt < 4; ++nt) {
    int och = nt * 16 + nloc;
    if (och < OFFC_) {
      float bias = off_b[och];
      #pragma unroll
      for (int r = 0; r < 8; ++r) {
        int m = r + hm * 8;
        offs[(((size_t)b * OFFC_ + och) * H_ + h) * W_ + (wb + m)] = acc[nt][r] + bias;
      }
    }
  }
}

// ---------- stage 3: deformable conv (gather + GEMM), grid.y = branch ----------
__global__ __launch_bounds__(32) void deform_gemm_kernel(
    const __bf16* __restrict__ xh, const float* __restrict__ offs,
    const uint32_t* __restrict__ BdAll, __bf16* __restrict__ ybr) {
  __shared__ __bf16 As[16 * AS_STRIDE];
  __shared__ float  offbuf[18 * 16];
  int branch = blockIdx.y;
  int dil = branch + 1, pad = dil;
  int tile = blockIdx.x;
  int wb = (tile & 7) * 16, h = (tile >> 3) & 127, b = tile >> 10;
  int lane = threadIdx.x;
  const uint32_t* Bp = BdAll + (size_t)branch * 18 * 4 * 32 * 8;
  __builtin_prefetch(Bp, 0, 3);
#if HAS_ASYNC_LDS
  // 18 planes x 16 pixels of offsets, 288 dwords: 9 async b32 per lane
  #pragma unroll
  for (int p = 0; p < 9; ++p) {
    int e = p * 32 + lane;
    int plane = e >> 4, mm = e & 15;
    const float* gsrc =
        offs + (((size_t)b * OFFC_ + branch * 18 + plane) * H_ + h) * W_ + wb + mm;
    __builtin_amdgcn_global_load_async_to_lds_b32(GLOBAL_AS(gsrc),
                                                  LDS_AS(offbuf + plane * 16 + mm), 0, 0);
  }
  WAIT_ASYNC();
#else
  for (int p = 0; p < 18; ++p)
    if (lane < 16)
      offbuf[p * 16 + lane] =
          offs[(((size_t)b * OFFC_ + branch * 18 + p) * H_ + h) * W_ + wb + lane];
#endif
  __syncthreads();
  int c2 = lane * 2;
  for (int m = 0; m < 16; ++m) {
    int w = wb + m;
    #pragma unroll
    for (int kk = 0; kk < KK_; ++kk) {
      float dy = offbuf[(kk * 2 + 0) * 16 + m];
      float dx = offbuf[(kk * 2 + 1) * 16 + m];
      float py = (float)(h - pad + (kk / 3) * dil) + dy;
      float px = (float)(w - pad + (kk % 3) * dil) + dx;
      float y0 = floorf(py), x0 = floorf(px);
      float wy1 = py - y0, wx1 = px - x0;
      int iy0 = (int)y0, ix0 = (int)x0;
      float a0 = 0.f, a1 = 0.f;
      #pragma unroll
      for (int cr = 0; cr < 4; ++cr) {
        int iy = iy0 + (cr >> 1), ix = ix0 + (cr & 1);
        float wgt = ((cr >> 1) ? wy1 : 1.f - wy1) * ((cr & 1) ? wx1 : 1.f - wx1);
        if (iy >= 0 && iy < H_ && ix >= 0 && ix < W_) {
          BF2 pv;
          pv.u = *(const uint32_t*)(xh + (((size_t)b * H_ + iy) * W_ + ix) * C_ + c2);
          a0 += wgt * (float)pv.h[0];
          a1 += wgt * (float)pv.h[1];
        }
      }
      BF2 pk; pk.h[0] = (__bf16)a0; pk.h[1] = (__bf16)a1;
      *(uint32_t*)(As + m * AS_STRIDE + kk * 64 + c2) = pk.u;
    }
  }
  __syncthreads();
  v8f acc[4] = {};
  for (int ks = 0; ks < 18; ++ks) {
    v16bf a = load_a_frag_lds(As, AS_STRIDE, ks, lane);
    #pragma unroll
    for (int nt = 0; nt < 4; ++nt)
      acc[nt] = wmma_bf16(a, load_b_frag(Bp, ks, nt, 4, lane), acc[nt]);
  }
  int nloc = lane & 15, hm = lane >> 4;
  size_t pixbase = ((size_t)b * H_ + h) * W_ + wb;
  #pragma unroll
  for (int nt = 0; nt < 4; ++nt) {
    int och = nt * 16 + nloc;
    #pragma unroll
    for (int r = 0; r < 8; ++r) {
      int m = r + hm * 8;
      ybr[(pixbase + m) * (3 * C_) + branch * C_ + och] = (__bf16)acc[nt][r];
    }
  }
}

// ---------- stage 4: fuse 1x1 (K=192) + residual ----------
__global__ __launch_bounds__(32) void fuse_kernel(
    const __bf16* __restrict__ ybr, const uint32_t* __restrict__ Bf,
    const float* __restrict__ out0, float* __restrict__ out1, __bf16* __restrict__ out1h) {
  int tile = blockIdx.x;
  int wb = (tile & 7) * 16, h = (tile >> 3) & 127, b = tile >> 10;
  int lane = threadIdx.x;
  int m = lane & 15, half = lane >> 4;
  size_t pixbase = ((size_t)b * H_ + h) * W_ + wb;
  v8f acc[4] = {};
  for (int ks = 0; ks < 6; ++ks) {
    const __bf16* arow = ybr + (pixbase + m) * (3 * C_) + ks * 32 + half * 8;
    AFrag u;
    u.q[0] = *(const u32x4*)arow;
    u.q[1] = *(const u32x4*)(arow + 16);
    #pragma unroll
    for (int nt = 0; nt < 4; ++nt)
      acc[nt] = wmma_bf16(u.v, load_b_frag(Bf, ks, nt, 4, lane), acc[nt]);
  }
  int nloc = lane & 15, hm = lane >> 4;
  #pragma unroll
  for (int nt = 0; nt < 4; ++nt) {
    int och = nt * 16 + nloc;
    #pragma unroll
    for (int r = 0; r < 8; ++r) {
      int mm = r + hm * 8;
      size_t idx = (((size_t)b * C_ + och) * H_ + h) * W_ + (wb + mm);
      float v = out0[idx] + acc[nt][r];
      out1[idx] = v;
      out1h[(pixbase + mm) * C_ + och] = (__bf16)v;
    }
  }
}

// ---------- stage 5: global average pool + SE gate ----------
__global__ void pool_kernel(const float* __restrict__ out1, float* __restrict__ pooled) {
  __shared__ float red[256];
  int bc = blockIdx.x, tid = threadIdx.x;
  const float* p = out1 + (size_t)bc * (H_ * W_);
  float s = 0.f;
  for (int i = tid; i < H_ * W_; i += 256) s += p[i];
  red[tid] = s;
  __syncthreads();
  for (int st = 128; st > 0; st >>= 1) {
    if (tid < st) red[tid] += red[tid + st];
    __syncthreads();
  }
  if (tid == 0) pooled[bc] = red[0] * (1.f / (H_ * W_));
}

__global__ void se_kernel(const float* __restrict__ pooled,
                          const float* __restrict__ se_w1, const float* __restrict__ se_w2,
                          float* __restrict__ gate) {
  __shared__ float t1[B_][4];
  int tid = threadIdx.x;
  int b = tid >> 6, j = tid & 63;
  if (j < 4) {
    float s = 0.f;
    for (int c = 0; c < C_; ++c) s += pooled[b * C_ + c] * se_w1[j * C_ + c];
    t1[b][j] = s / (1.f + __expf(-s));
  }
  __syncthreads();
  int c = tid & 63;
  float s = 0.f;
  #pragma unroll
  for (int jj = 0; jj < 4; ++jj) s += t1[b][jj] * se_w2[c * 4 + jj];
  gate[b * C_ + c] = 1.f / (1.f + __expf(-s));
}

// ---------- stage 6: SE scale + FFN (GEMM1 -> SiLU -> GEMM2) + residual ----------
__global__ __launch_bounds__(32) void ffn_kernel(
    const float* __restrict__ out1, const __bf16* __restrict__ out1h,
    const float* __restrict__ gate,
    const uint32_t* __restrict__ Bw1, const float* __restrict__ b1,
    const uint32_t* __restrict__ Bw2, const float* __restrict__ b2,
    float* __restrict__ outF) {
  __shared__ __bf16 Hs[16 * HS_STRIDE];
  int tile = blockIdx.x;
  int wb = (tile & 7) * 16, h = (tile >> 3) & 127, b = tile >> 10;
  int lane = threadIdx.x;
  int m = lane & 15, half = lane >> 4;
  size_t pixbase = ((size_t)b * H_ + h) * W_ + wb;
  const float* grow = gate + b * C_;
  __builtin_prefetch(Bw1, 0, 3);
  __builtin_prefetch(Bw2, 0, 3);

  v8f hacc[8] = {};
  for (int ks = 0; ks < 2; ++ks) {
    const __bf16* arow = out1h + (pixbase + m) * C_ + ks * 32 + half * 8;
    AFrag u;
    u.q[0] = *(const u32x4*)arow;
    u.q[1] = *(const u32x4*)(arow + 16);
    int kb = ks * 32 + half * 8;
    #pragma unroll
    for (int i = 0; i < 8; ++i) {
      u.h[i]     = (__bf16)((float)u.h[i]     * grow[kb + i]);
      u.h[i + 8] = (__bf16)((float)u.h[i + 8] * grow[kb + 16 + i]);
    }
    #pragma unroll
    for (int nt = 0; nt < 8; ++nt)
      hacc[nt] = wmma_bf16(u.v, load_b_frag(Bw1, ks, nt, 8, lane), hacc[nt]);
  }
  int nloc = lane & 15, hm = lane >> 4;
  #pragma unroll
  for (int nt = 0; nt < 8; ++nt) {
    int och = nt * 16 + nloc;
    float bias = b1[och];
    #pragma unroll
    for (int r = 0; r < 8; ++r) {
      int mm = r + hm * 8;
      float v = hacc[nt][r] + bias;
      v = v / (1.f + __expf(-v));
      Hs[mm * HS_STRIDE + och] = (__bf16)v;
    }
  }
  __syncthreads();
  v8f acc2[4] = {};
  for (int ks = 0; ks < 4; ++ks) {
    v16bf a = load_a_frag_lds(Hs, HS_STRIDE, ks, lane);
    #pragma unroll
    for (int nt = 0; nt < 4; ++nt)
      acc2[nt] = wmma_bf16(a, load_b_frag(Bw2, ks, nt, 4, lane), acc2[nt]);
  }
  #pragma unroll
  for (int nt = 0; nt < 4; ++nt) {
    int och = nt * 16 + nloc;
    float g = gate[b * C_ + och];
    float bias = b2[och];
    #pragma unroll
    for (int r = 0; r < 8; ++r) {
      int mm = r + hm * 8;
      size_t idx = (((size_t)b * C_ + och) * H_ + h) * W_ + (wb + mm);
      outF[idx] = out1[idx] * g + acc2[nt][r] + bias;
    }
  }
}

// ---------- host ----------
extern "C" void kernel_launch(void* const* d_in, const int* in_sizes, int n_in,
                              void* d_out, int out_size, void* d_ws, size_t ws_size,
                              hipStream_t stream) {
  const float* x        = (const float*)d_in[0];
  const float* bn_gamma = (const float*)d_in[1];
  const float* bn_beta  = (const float*)d_in[2];
  const float* bn_mean  = (const float*)d_in[3];
  const float* bn_var   = (const float*)d_in[4];
  const float* resscale = (const float*)d_in[5];
  const float* off_w    = (const float*)d_in[6];
  const float* off_b    = (const float*)d_in[7];
  const float* dw[3]    = {(const float*)d_in[8], (const float*)d_in[9], (const float*)d_in[10]};
  const float* fuse_w   = (const float*)d_in[11];
  const float* se_w1    = (const float*)d_in[12];
  const float* se_w2    = (const float*)d_in[13];
  const float* ffn_w1   = (const float*)d_in[14];
  const float* ffn_b1   = (const float*)d_in[15];
  const float* ffn_w2   = (const float*)d_in[16];
  const float* ffn_b2   = (const float*)d_in[17];
  float* outF = (float*)d_out;

  // carve workspace (256B aligned)
  char* ws = (char*)d_ws;
  size_t o = 0;
  auto carve = [&](size_t bytes) -> char* {
    o = (o + 255) & ~(size_t)255;
    char* p = ws + o;
    o += bytes;
    return p;
  };
  float*    out0   = (float*)   carve((size_t)NPIX * C_ * 4);
  __bf16*   out0h  = (__bf16*)  carve((size_t)NPIX * C_ * 2);
  float*    offs   = (float*)   carve((size_t)B_ * OFFC_ * H_ * W_ * 4);
  __bf16*   ybr    = (__bf16*)  carve((size_t)NPIX * 3 * C_ * 2);
  float*    out1   = (float*)   carve((size_t)NPIX * C_ * 4);
  __bf16*   out1h  = (__bf16*)  carve((size_t)NPIX * C_ * 2);
  float*    pooled = (float*)   carve(B_ * C_ * 4);
  float*    gate   = (float*)   carve(B_ * C_ * 4);
  uint32_t* Boff   = (uint32_t*)carve((size_t)18 * 4 * 256 * 4);
  uint32_t* Bd     = (uint32_t*)carve((size_t)3 * 18 * 4 * 256 * 4);
  uint32_t* Bfuse  = (uint32_t*)carve((size_t)6 * 4 * 256 * 4);
  uint32_t* Bffn1  = (uint32_t*)carve((size_t)2 * 8 * 256 * 4);
  uint32_t* Bffn2  = (uint32_t*)carve((size_t)4 * 4 * 256 * 4);

  // weight packing
  pack_b_kernel<<<72, 256, 0, stream>>>(off_w, Boff, 0, 576, OFFC_, 18, 4);
  for (int br = 0; br < 3; ++br)
    pack_b_kernel<<<72, 256, 0, stream>>>(dw[br], Bd + (size_t)br * 18 * 4 * 256, 0, 576, C_, 18, 4);
  pack_b_kernel<<<24, 256, 0, stream>>>(fuse_w,  Bfuse, 1, 192, C_,     6, 4);
  pack_b_kernel<<<16, 256, 0, stream>>>(ffn_w1,  Bffn1, 1,  64, 2 * C_, 2, 8);
  pack_b_kernel<<<16, 256, 0, stream>>>(ffn_w2,  Bffn2, 1, 128, C_,     4, 4);

  // pipeline
  bn_silu_kernel<<<(NPIX * C_) / 256, 256, 0, stream>>>(x, bn_gamma, bn_beta, bn_mean, bn_var,
                                                        resscale, out0, out0h);
  offset_conv_kernel<<<NTILE, 32, 0, stream>>>(out0h, Boff, off_b, offs);
  deform_gemm_kernel<<<dim3(NTILE, 3), 32, 0, stream>>>(out0h, offs, Bd, ybr);
  fuse_kernel<<<NTILE, 32, 0, stream>>>(ybr, Bfuse, out0, out1, out1h);
  pool_kernel<<<B_ * C_, 256, 0, stream>>>(out1, pooled);
  se_kernel<<<1, 256, 0, stream>>>(pooled, se_w1, se_w2, gate);
  ffn_kernel<<<NTILE, 32, 0, stream>>>(out1, out1h, gate, Bffn1, ffn_b1, Bffn2, ffn_b2, outF);
}
